// GNN_MNIST_65489661330093
// MI455X (gfx1250) — matrix operand order repaired and checked by reading, not compiled, not run on hardware
//
#include <hip/hip_runtime.h>
#include <hip/hip_bf16.h>
#include <hip/hip_fp16.h>
#include <stdint.h>

typedef __attribute__((ext_vector_type(16))) _Float16 v16h;
typedef __attribute__((ext_vector_type(8)))  _Float16 v8h;
typedef __attribute__((ext_vector_type(8)))  float    v8f;

// ---------------------------------------------------------------- utilities
__global__ void fill_f32(float* __restrict__ p, float v, int n) {
    int i = blockIdx.x * blockDim.x + threadIdx.x;
    if (i < n) p[i] = v;
}

// deg[d] += 1 for every edge (deg pre-filled with 1.0 for the self-loop)
__global__ void deg_count(const int* __restrict__ dst, float* __restrict__ deg, int E) {
    int e = blockIdx.x * blockDim.x + threadIdx.x;
    if (e < E) atomicAdd(&deg[dst[e]], 1.0f);
}

__global__ void deg_to_dinv(float* __restrict__ d, int n) {
    int i = blockIdx.x * blockDim.x + threadIdx.x;
    if (i < n) d[i] = rsqrtf(d[i]);
}

__global__ void edge_norm(const int* __restrict__ src, const int* __restrict__ dst,
                          const float* __restrict__ dinv, float* __restrict__ norm, int E) {
    int e = blockIdx.x * blockDim.x + threadIdx.x;
    if (e < E) norm[e] = dinv[src[e]] * dinv[dst[e]];
}

// ------------------------------------------------- layer 1 GEMM (K=3, scalar)
__global__ void gemm_in3(const float* __restrict__ x, const float* __restrict__ W,
                         float* __restrict__ t, int N) {
    int gid = blockIdx.x * blockDim.x + threadIdx.x;
    if (gid >= N * 64) return;
    int i = gid >> 6, j = gid & 63;
    float x0 = x[3 * i], x1 = x[3 * i + 1], x2 = x[3 * i + 2];
    t[gid] = fmaf(x0, W[j], fmaf(x1, W[64 + j], x2 * W[128 + j]));
}

// ---------------- pack fp32 W[K,Ko] into f16 WMMA B-fragment layout ----------
// Fragment element (kstep s, colTile ct, lane, h) = W[32*s + 16*(lane>>4) + h][16*ct + (lane&15)]
// stored contiguously so each lane's 16 halves are one 32-byte run.
__global__ void pack_w(const float* __restrict__ W, _Float16* __restrict__ Wp,
                       int K, int Ko) {
    int colTiles = Ko >> 4;
    int total = (K >> 5) * colTiles * 32 * 16;
    int gid = blockIdx.x * blockDim.x + threadIdx.x;
    if (gid >= total) return;
    int h    = gid & 15;
    int lane = (gid >> 4) & 31;
    int rest = gid >> 9;
    int ct   = rest % colTiles;
    int s    = rest / colTiles;
    int n = lane & 15, hi = lane >> 4;
    int krow = (s << 5) + 16 * hi + h;
    Wp[gid] = (_Float16)W[(size_t)krow * Ko + (ct << 4) + n];
}

// ---------------- WMMA GEMM: C[N,128] = Ah[N,K](f16) * Wp(packed f16) --------
// One wave computes a full 16-row x 128-col stripe: 8 accumulators, 8 back-to-
// back v_wmma per 32-wide k-step. A fragment = two contiguous v8h loads; B
// fragment = one contiguous 32B load per tile from the packed weights.
// Requires K % 32 == 0; launch must cover full 16-row tiles (N % 16 == 0 here).
__global__ void wmma_gemm128(const _Float16* __restrict__ Ah,
                             const _Float16* __restrict__ Wp,
                             float* __restrict__ C, int N, int K) {
    int wave = (blockIdx.x * blockDim.x + threadIdx.x) >> 5;
    int lane = threadIdx.x & 31;
    if (wave * 16 >= N) return;             // wave-uniform: EXEC stays all-ones

    const int m  = lane & 15;
    const int hi = lane >> 4;
    const size_t rowoff = (size_t)(wave * 16 + m) * K;

    v8f acc[8] = {};
    const int ksteps = K >> 5;
    for (int s = 0; s < ksteps; ++s) {
        const int kb = s << 5;
        const v8h a0 = *(const v8h*)(Ah + rowoff + kb + 16 * hi);
        const v8h a1 = *(const v8h*)(Ah + rowoff + kb + 16 * hi + 8);
        const v16h a = __builtin_shufflevector(a0, a1,
            0, 1, 2, 3, 4, 5, 6, 7, 8, 9, 10, 11, 12, 13, 14, 15);
        const _Float16* wbase = Wp + (size_t)s * (8 * 32 * 16) + lane * 16;
#pragma unroll
        for (int ct = 0; ct < 8; ++ct) {
            const v16h b = *(const v16h*)(wbase + (size_t)ct * (32 * 16));
            acc[ct] = __builtin_amdgcn_wmma_f32_16x16x32_f16(
                /*neg_a=*/false, a, /*neg_b=*/false, b,
                /*c_mod=*/(short)0, acc[ct], /*reuse_a=*/false, /*reuse_b=*/false);
        }
    }

    // C layout: acc[ct][r] -> row wave*16 + 8*hi + r, col ct*16 + m
    const int mbase = wave * 16 + 8 * hi;
#pragma unroll
    for (int ct = 0; ct < 8; ++ct) {
        float* crow = C + (size_t)mbase * 128 + (ct << 4) + m;
#pragma unroll
        for (int r = 0; r < 8; ++r) crow[(size_t)r * 128] = acc[ct][r];
    }
}

// ----------------------------- edge scatter: agg[dst] += t[src] * norm (atomics)
// One thread per (edge, 4 features). fshift = log2(F).
__global__ void scatter_edges(const float* __restrict__ t, const float* __restrict__ norm,
                              const int* __restrict__ src, const int* __restrict__ dst,
                              float* __restrict__ agg, int E, int fshift) {
    int gid = blockIdx.x * blockDim.x + threadIdx.x;
    int q = fshift - 2;                      // log2(F/4)
    if (gid >= (E << q)) return;
    int e = gid >> q;
    int f = (gid & ((1 << q) - 1)) << 2;
    int s = src[e], d = dst[e];
    float nv = norm[e];
    const float4 v = *(const float4*)(t + ((size_t)s << fshift) + f);
    float* o = agg + ((size_t)d << fshift) + f;
    atomicAdd(o + 0, v.x * nv);
    atomicAdd(o + 1, v.y * nv);
    atomicAdd(o + 2, v.z * nv);
    atomicAdd(o + 3, v.w * nv);
}

// ---- fused self-loop + bias + ReLU: h = relu(agg + t*dinv^2 + bias),
// emitting both fp32 (for pooling) and f16 (A operand of the next WMMA GEMM).
__global__ void finish_layer(const float* __restrict__ t, const float* __restrict__ agg,
                             const float* __restrict__ dinv, const float* __restrict__ bias,
                             float* __restrict__ h32, _Float16* __restrict__ h16,
                             int N, int fshift) {
    int gid = blockIdx.x * blockDim.x + threadIdx.x;
    if (gid >= (N << fshift)) return;
    int i = gid >> fshift;
    int f = gid & ((1 << fshift) - 1);
    float di = dinv[i];
    float v = agg[gid] + t[gid] * di * di + bias[f];
    v = v > 0.0f ? v : 0.0f;
    h32[gid] = v;
    h16[gid] = (_Float16)v;
}

// ------------------------------------------------------------------- pooling
__global__ void pool_cnt(const int* __restrict__ batch, float* __restrict__ cnt, int N) {
    int i = blockIdx.x * blockDim.x + threadIdx.x;
    if (i < N) atomicAdd(&cnt[batch[i]], 1.0f);
}

__global__ void pool_sum(const float* __restrict__ h, const int* __restrict__ batch,
                         float* __restrict__ sums, int N) {
    int gid = blockIdx.x * blockDim.x + threadIdx.x;
    if (gid >= N * 128) return;
    int i = gid >> 7, f = gid & 127;
    atomicAdd(&sums[(batch[i] << 7) + f], h[gid]);
}

// ------------------------------------------- MLP head (single block, via LDS)
__global__ void head_mlp(const float* __restrict__ sums, const float* __restrict__ cnt,
                         const float* __restrict__ Wf1, const float* __restrict__ bf1,
                         const float* __restrict__ Wf2, const float* __restrict__ bf2,
                         float* __restrict__ out) {
    __shared__ float pooled[64 * 128];
    __shared__ float fc1[64 * 64];
    int t = threadIdx.x;
    for (int idx = t; idx < 64 * 128; idx += blockDim.x) {
        int g = idx >> 7;
        float c = cnt[g]; c = c < 1.0f ? 1.0f : c;
        pooled[idx] = sums[idx] / c;
    }
    __syncthreads();
    for (int idx = t; idx < 64 * 64; idx += blockDim.x) {
        int g = idx >> 6, j = idx & 63;
        float s = bf1[j];
        const float* pg = pooled + (g << 7);
#pragma unroll 4
        for (int f = 0; f < 128; ++f) s = fmaf(pg[f], Wf1[f * 64 + j], s);
        fc1[idx] = s > 0.0f ? s : 0.0f;
    }
    __syncthreads();
    for (int idx = t; idx < 64 * 10; idx += blockDim.x) {
        int g = idx / 10, o = idx - g * 10;
        float s = bf2[o];
        const float* fg = fc1 + (g << 6);
#pragma unroll 4
        for (int j = 0; j < 64; ++j) s = fmaf(fg[j], Wf2[j * 10 + o], s);
        out[idx] = s;
    }
}

// -------------------------------------------------------------------- launch
extern "C" void kernel_launch(void* const* d_in, const int* in_sizes, int n_in,
                              void* d_out, int out_size, void* d_ws, size_t ws_size,
                              hipStream_t stream) {
    const float* x     = (const float*)d_in[0];
    const int*   ei    = (const int*)d_in[1];
    const int*   batch = (const int*)d_in[2];
    const float* W1 = (const float*)d_in[3];  const float* b1  = (const float*)d_in[4];
    const float* W2 = (const float*)d_in[5];  const float* b2  = (const float*)d_in[6];
    const float* W3 = (const float*)d_in[7];  const float* b3  = (const float*)d_in[8];
    const float* Wf1 = (const float*)d_in[9]; const float* bf1 = (const float*)d_in[10];
    const float* Wf2 = (const float*)d_in[11];const float* bf2 = (const float*)d_in[12];
    float* out = (float*)d_out;

    const int N = in_sizes[0] / 3;     // 50000
    const int E = in_sizes[1] / 2;     // 800000
    const int* src = ei;
    const int* dst = ei + E;

    // workspace layout (64B aligned slices)
    uintptr_t base = (uintptr_t)d_ws;
    auto take = [&](size_t bytes) {
        base = (base + 63) & ~(uintptr_t)63;
        uintptr_t p = base;
        base += bytes;
        return (void*)p;
    };
    float*     dinv = (float*)take((size_t)N * 4);
    float*     norm = (float*)take((size_t)E * 4);
    float*     buf0 = (float*)take((size_t)N * 128 * 4);
    float*     buf1 = (float*)take((size_t)N * 128 * 4);
    float*     buf2 = (float*)take((size_t)N * 128 * 4);
    _Float16*  bufh = (_Float16*)take((size_t)N * 128 * 2);
    _Float16*  Wp2  = (_Float16*)take((size_t)64 * 128 * 2);
    _Float16*  Wp3  = (_Float16*)take((size_t)128 * 128 * 2);
    float*     psum = (float*)take(64 * 128 * 4);
    float*     pcnt = (float*)take(64 * 4);

    auto nb = [](long long n, int bs) { return (unsigned)((n + bs - 1) / bs); };

    // ---- pack weights into f16 WMMA fragment layout (reused by 3125 waves)
    pack_w<<<nb(64 * 128, 256), 256, 0, stream>>>(W2, Wp2, 64, 128);
    pack_w<<<nb(128 * 128, 256), 256, 0, stream>>>(W3, Wp3, 128, 128);

    // ---- degrees (self-loop = 1) -> dinv -> per-edge norm
    fill_f32   <<<nb(N, 256), 256, 0, stream>>>(dinv, 1.0f, N);
    deg_count  <<<nb(E, 256), 256, 0, stream>>>(dst, dinv, E);
    deg_to_dinv<<<nb(N, 256), 256, 0, stream>>>(dinv, N);
    edge_norm  <<<nb(E, 256), 256, 0, stream>>>(src, dst, dinv, norm, E);

    // ---- layer 1: t = x@W1 (K=3 scalar), scatter, fused self-loop+bias+relu
    gemm_in3     <<<nb((long long)N * 64, 256), 256, 0, stream>>>(x, W1, buf0, N);
    fill_f32     <<<nb((long long)N * 64, 256), 256, 0, stream>>>(buf1, 0.0f, N * 64);
    scatter_edges<<<nb((long long)E * 16, 256), 256, 0, stream>>>(buf0, norm, src, dst, buf1, E, 6);
    finish_layer <<<nb((long long)N * 64, 256), 256, 0, stream>>>(buf0, buf1, dinv, b1, buf1, bufh, N, 6);

    // ---- layer 2: t = h1[N,64](f16) @ W2 -> buf0[N,128] via WMMA
    {
        long long thr = (long long)((N + 15) / 16) * 32;
        wmma_gemm128<<<nb(thr, 256), 256, 0, stream>>>(bufh, Wp2, buf0, N, 64);
    }
    fill_f32     <<<nb((long long)N * 128, 256), 256, 0, stream>>>(buf2, 0.0f, N * 128);
    scatter_edges<<<nb((long long)E * 32, 256), 256, 0, stream>>>(buf0, norm, src, dst, buf2, E, 7);
    finish_layer <<<nb((long long)N * 128, 256), 256, 0, stream>>>(buf0, buf2, dinv, b2, buf2, bufh, N, 7);

    // ---- layer 3: t = h2[N,128](f16) @ W3 -> buf0[N,128] via WMMA
    {
        long long thr = (long long)((N + 15) / 16) * 32;
        wmma_gemm128<<<nb(thr, 256), 256, 0, stream>>>(bufh, Wp3, buf0, N, 128);
    }
    fill_f32     <<<nb((long long)N * 128, 256), 256, 0, stream>>>(buf1, 0.0f, N * 128);
    scatter_edges<<<nb((long long)E * 32, 256), 256, 0, stream>>>(buf0, norm, src, dst, buf1, E, 7);
    finish_layer <<<nb((long long)N * 128, 256), 256, 0, stream>>>(buf0, buf1, dinv, b3, buf1, bufh, N, 7);

    // ---- global mean pool + MLP head
    fill_f32<<<nb(64 * 128, 256), 256, 0, stream>>>(psum, 0.0f, 64 * 128);
    fill_f32<<<1, 64, 0, stream>>>(pcnt, 0.0f, 64);
    pool_cnt<<<nb(N, 256), 256, 0, stream>>>(batch, pcnt, N);
    pool_sum<<<nb((long long)N * 128, 256), 256, 0, stream>>>(buf1, batch, psum, N);
    head_mlp<<<1, 256, 0, stream>>>(psum, pcnt, Wf1, bf1, Wf2, bf2, out);
}
